// Capsule_Net_64836826300566
// MI455X (gfx1250) — matrix operand order repaired and checked by reading, not compiled, not run on hardware
//
#include <hip/hip_runtime.h>
#include <hip/hip_bf16.h>
#include <math.h>

typedef __bf16 bf16;
typedef __attribute__((ext_vector_type(16))) __bf16 v16bf;
typedef __attribute__((ext_vector_type(8)))  float  v8f;

#define WMMA_BF16(A, B, C) \
  __builtin_amdgcn_wmma_f32_16x16x32_bf16(false, (A), false, (B), (short)0, (C), false, false)

// ---------------------------------------------------------------------------
// Fragment loaders (wave32), contiguous k = 16*(lane/16) + element for both A and B.
// A (16x32, MxK): lane L<16 -> M=L, K in {0..15}; lane>=16 -> M=L-16,
//   K in {16..31}. Caller pre-offsets the row pointer by k0 + (hi?16:0).
__device__ inline v16bf load_frag_A16(const bf16* __restrict__ p) {
  v16bf a;
#pragma unroll
  for (int e = 0; e < 8; ++e) { a[e] = p[e]; a[8 + e] = p[8 + e]; }
  return a;
}
// B (32x16, KxN) from weights stored row-major [N][K]: lane L -> N=L&15,
//   K = (L<16?0:16)+2j+{0,1}. Caller pre-offsets row pointer by k0 + (hi?16:0).
__device__ inline v16bf load_frag_B16(const bf16* __restrict__ p) {
  v16bf b;
#pragma unroll
  for (int e = 0; e < 16; ++e) b[e] = p[e];
  return b;
}

// ---------------------------------------------------------------------------
// Weight prep kernels (run every launch; deterministic, no caching).
__global__ void prep_w1(const float* __restrict__ w, bf16* __restrict__ out) {
  int i = blockIdx.x * 256 + threadIdx.x;              // 256*96
  if (i < 256 * 96) {
    int n = i / 96, k = i % 96;
    out[i] = (k < 81) ? (bf16)w[n * 81 + k] : (bf16)0.0f;
  }
}
__global__ void prep_w2(const float* __restrict__ w, bf16* __restrict__ out) {
  int i = blockIdx.x * 256 + threadIdx.x;              // 256*20736
  if (i < 256 * 20736) {
    int n = i / 20736, rem = i % 20736;
    int s = rem / 256, c = rem % 256;                  // k=(ky*9+kx)*256+c
    out[i] = (bf16)w[(size_t)n * 20736 + c * 81 + s];
  }
}
__global__ void cvt_f32_bf16(const float* __restrict__ in, bf16* __restrict__ out, int n) {
  int i = blockIdx.x * 256 + threadIdx.x;
  if (i < n) out[i] = (bf16)in[i];
}

// ---------------------------------------------------------------------------
// Zero-padded bf16 im2col for conv1: A1[m][k], m in [0,102400), k in [0,96)
// (k>=81 padded to zero). Keeps the GEMM kernel branch-free (EXEC all-ones).
__global__ void im2col1(const float* __restrict__ x, bf16* __restrict__ A1) {
  int idx = blockIdx.x * 256 + threadIdx.x;            // 102400*96
  if (idx < 102400 * 96) {
    int m = idx / 96, k = idx % 96;
    bf16 v = (bf16)0.0f;
    if (k < 81) {
      int b = m / 400, r = m % 400, oy = r / 20, ox = r % 20;
      int ky = k / 9, kx = k % 9;
      v = (bf16)x[(size_t)b * 784 + (oy + ky) * 28 + ox + kx];
    }
    A1[idx] = v;
  }
}

// ---------------------------------------------------------------------------
// conv1 as GEMM: M=102400, K=96, N=256.  A1 bf16 [M][96], B = w1b [N][96] bf16.
// Output h1 NHWC bf16: h1[((b*20+y)*20+x)*256 + n], bias fused.
__global__ void conv1_wmma(const bf16* __restrict__ A1, const bf16* __restrict__ w1b,
                           const float* __restrict__ bias, bf16* __restrict__ h1) {
  int lane = threadIdx.x, l15 = lane & 15, hi = lane >> 4;
  int m0 = blockIdx.x * 16, n0 = blockIdx.y * 16;
  const bf16* arow = A1 + (size_t)(m0 + l15) * 96 + (hi ? 16 : 0);
  const bf16* brow = w1b + (size_t)(n0 + l15) * 96 + (hi ? 16 : 0);
  v8f acc = {};
#pragma unroll
  for (int kc = 0; kc < 3; ++kc) {
    v16bf a  = load_frag_A16(arow + kc * 32);
    v16bf bb = load_frag_B16(brow + kc * 32);
    acc = WMMA_BF16(a, bb, acc);
  }
  int n = n0 + l15;
  float bv = bias[n];
#pragma unroll
  for (int j = 0; j < 8; ++j) {
    int mm = m0 + j + hi * 8;                          // 16 | 400 -> one image per tile
    int b2 = mm / 400, r2 = mm % 400, yy = r2 / 20, xx = r2 % 20;
    h1[(((size_t)b2 * 20 + yy) * 20 + xx) * 256 + n] = (bf16)(acc[j] + bv);
  }
}

// ---------------------------------------------------------------------------
// pconv (stride 2) as GEMM: M=B*36=9216, K=81*256=20736, N=256.  Dominant kernel.
// 2x2 register blocking: each wave owns a 32x32 output block (4 accumulators),
// sharing A fragments across 2 N-tiles and B fragments across 2 M-tiles
// -> 16 FLOP/byte of fragment traffic (2x the unblocked version).
// A from h1 NHWC bf16 (contiguous 32-channel K-chunks), B = w2b [N][K] bf16.
// Output h2 fp32, NCHW-flat: h2[b*9216 + n*36 + oy*6+ox], bias fused.
__global__ void pconv_wmma(const bf16* __restrict__ h1, const bf16* __restrict__ w2b,
                           const float* __restrict__ bias, float* __restrict__ h2) {
  int lane = threadIdx.x, l15 = lane & 15, hi = lane >> 4;
  int m0 = blockIdx.x * 32, n0 = blockIdx.y * 32;
  int bA[2], spA[2];                                   // per-M-tile lane row decomposition
#pragma unroll
  for (int t = 0; t < 2; ++t) {
    int m = m0 + 16 * t + l15;
    bA[t] = m / 36;
    int r = m % 36;
    spA[t] = (r / 6) * 2 * 20 + (r % 6) * 2;           // (oy*2)*20 + ox*2
  }
  const bf16* br0 = w2b + (size_t)(n0 + l15) * 20736 + (hi ? 16 : 0);
  const bf16* br1 = w2b + (size_t)(n0 + 16 + l15) * 20736 + (hi ? 16 : 0);
  v8f acc00 = {}, acc01 = {}, acc10 = {}, acc11 = {};
  for (int s = 0; s < 81; ++s) {
    int ky = s / 9, kx = s % 9;
    const bf16* ar0 = h1 + (((size_t)bA[0] * 400) + spA[0] + ky * 20 + kx) * 256 + (hi ? 16 : 0);
    const bf16* ar1 = h1 + (((size_t)bA[1] * 400) + spA[1] + ky * 20 + kx) * 256 + (hi ? 16 : 0);
    // next filter tap is +256 elements (contiguous) for kx<8: prefetch it
    __builtin_prefetch(ar0 + 256, 0, 1);
    __builtin_prefetch(ar1 + 256, 0, 1);
    const bf16* b0s = br0 + s * 256;
    const bf16* b1s = br1 + s * 256;
#pragma unroll
    for (int cc = 0; cc < 8; ++cc) {                   // 8 chunks of 32 channels
      v16bf a0 = load_frag_A16(ar0 + cc * 32);
      v16bf a1 = load_frag_A16(ar1 + cc * 32);
      v16bf b0 = load_frag_B16(b0s + cc * 32);
      v16bf b1 = load_frag_B16(b1s + cc * 32);
      acc00 = WMMA_BF16(a0, b0, acc00);
      acc01 = WMMA_BF16(a0, b1, acc01);
      acc10 = WMMA_BF16(a1, b0, acc10);
      acc11 = WMMA_BF16(a1, b1, acc11);
    }
  }
  int nc0 = n0 + l15, nc1 = n0 + 16 + l15;
  float bv0 = bias[nc0], bv1 = bias[nc1];
#pragma unroll
  for (int j = 0; j < 8; ++j) {
    int mm0 = m0 + j + hi * 8;
    int mm1 = mm0 + 16;
    int ba0 = mm0 / 36, ra0 = mm0 % 36;
    int ba1 = mm1 / 36, ra1 = mm1 % 36;
    h2[(size_t)ba0 * 9216 + nc0 * 36 + ra0] = acc00[j] + bv0;
    h2[(size_t)ba0 * 9216 + nc1 * 36 + ra0] = acc01[j] + bv1;
    h2[(size_t)ba1 * 9216 + nc0 * 36 + ra1] = acc10[j] + bv0;
    h2[(size_t)ba1 * 9216 + nc1 * 36 + ra1] = acc11[j] + bv1;
  }
}

// ---------------------------------------------------------------------------
// caps = squash(h2.reshape(B,1152,8)); NCHW-flat h2 means caps elem d = h2[i*8+d].
__global__ void squash_caps(const float* __restrict__ h2, float* __restrict__ caps) {
  int i = blockIdx.x * 256 + threadIdx.x;               // B*1152
  if (i < 256 * 1152) {
    const float* p = h2 + (size_t)i * 8;
    float xv[8], n2 = 0.0f;
#pragma unroll
    for (int d = 0; d < 8; ++d) { xv[d] = p[d]; n2 += xv[d] * xv[d]; }
    float sc = sqrtf(n2) / (1.0f + n2);
#pragma unroll
    for (int d = 0; d < 8; ++d) caps[(size_t)i * 8 + d] = xv[d] * sc;
  }
}

// ---------------------------------------------------------------------------
// Dynamic routing, one block per (batch, out-capsule).  u[1152][16] lives in LDS
// (stride 17 to dodge bank conflicts); b_i is a scalar per input capsule since the
// softmax axis is i and the logit update is constant along n.  Deterministic tree
// reductions (no atomics).
#define U_STRIDE 17
__global__ void routing_kernel(const float* __restrict__ caps, const float* __restrict__ Wd,
                               float* __restrict__ vout) {
  int bo = blockIdx.x, b = bo / 10, o = bo % 10;
  int t = threadIdx.x;                                   // 256 threads
  extern __shared__ float sm[];
  float* u   = sm;                                       // 1152*17
  float* bb  = u + 1152 * U_STRIDE;                      // 1152
  float* cc  = bb + 1152;                                // 1152
  float* red = cc + 1152;                                // 256
  float* ssh = red + 256;                                // 16
  float* vsh = ssh + 16;                                 // 16

  for (int i = t; i < 1152; i += 256) {
    const float* cp = caps + ((size_t)b * 1152 + i) * 8;
    float cv[8];
#pragma unroll
    for (int d = 0; d < 8; ++d) cv[d] = cp[d];
    const float* wp = Wd + (((size_t)i * 10 + o) * 16) * 8;
#pragma unroll
    for (int n = 0; n < 16; ++n) {
      float acc = 0.0f;
#pragma unroll
      for (int d = 0; d < 8; ++d) acc += wp[n * 8 + d] * cv[d];
      u[i * U_STRIDE + n] = acc;
    }
    bb[i] = 0.0f;
  }
  __syncthreads();

  for (int r = 1; r <= 3; ++r) {
    // softmax over i
    float lm = -3.4e38f;
    for (int i = t; i < 1152; i += 256) lm = fmaxf(lm, bb[i]);
    red[t] = lm; __syncthreads();
    for (int s2 = 128; s2 > 0; s2 >>= 1) { if (t < s2) red[t] = fmaxf(red[t], red[t + s2]); __syncthreads(); }
    float mx = red[0]; __syncthreads();
    float ls = 0.0f;
    for (int i = t; i < 1152; i += 256) { float e = __expf(bb[i] - mx); cc[i] = e; ls += e; }
    red[t] = ls; __syncthreads();
    for (int s2 = 128; s2 > 0; s2 >>= 1) { if (t < s2) red[t] += red[t + s2]; __syncthreads(); }
    float inv = 1.0f / red[0]; __syncthreads();
    // s[n] = sum_i c_i * u[i][n]
    float ps[16];
#pragma unroll
    for (int n = 0; n < 16; ++n) ps[n] = 0.0f;
    for (int i = t; i < 1152; i += 256) {
      float ci = cc[i] * inv;
#pragma unroll
      for (int n = 0; n < 16; ++n) ps[n] += ci * u[i * U_STRIDE + n];
    }
    for (int n = 0; n < 16; ++n) {
      red[t] = ps[n]; __syncthreads();
      for (int s2 = 128; s2 > 0; s2 >>= 1) { if (t < s2) red[t] += red[t + s2]; __syncthreads(); }
      if (t == 0) ssh[n] = red[0];
      __syncthreads();
    }
    if (t == 0) {
      float n2 = 0.0f;
      for (int n = 0; n < 16; ++n) n2 += ssh[n] * ssh[n];
      float sc = sqrtf(n2) / (1.0f + n2);
      for (int n = 0; n < 16; ++n) vsh[n] = ssh[n] * sc;
    }
    __syncthreads();
    if (r != 3) {
      for (int i = t; i < 1152; i += 256) {
        float acc = 0.0f;
#pragma unroll
        for (int n = 0; n < 16; ++n) acc += vsh[n] * u[i * U_STRIDE + n];
        bb[i] += acc;
      }
      __syncthreads();
    }
  }
  if (t < 16) vout[((size_t)b * 10 + o) * 16 + t] = vsh[t];
}

// ---------------------------------------------------------------------------
// recon input mask (bf16) + one-hot output (fp32).
__global__ void mask_onehot(const float* __restrict__ vout, const int* __restrict__ label,
                            bf16* __restrict__ recin, float* __restrict__ onehot) {
  int idx = blockIdx.x * 256 + threadIdx.x;              // up to 256*160
  if (idx < 256 * 160) {
    int b = idx / 160, r = idx % 160, o = r / 16;
    recin[idx] = (o == label[b]) ? (bf16)vout[idx] : (bf16)0.0f;
  }
  if (idx < 2560) {
    int b = idx / 10, o = idx % 10;
    onehot[idx] = (label[b] == o) ? 1.0f : 0.0f;
  }
}

// ---------------------------------------------------------------------------
// Generic decoder GEMM: out[m][n] = act(sum_k A[m][k]*Wt[n][k] + bias[n]).
// A bf16 [M][K] row-major, Wt bf16 [N][K] row-major. act: 0=relu, 1=sigmoid.
__global__ void gemm_dec(const bf16* __restrict__ A, const bf16* __restrict__ Wt,
                         const float* __restrict__ bias, float* __restrict__ outf,
                         bf16* __restrict__ outbf, int K, int N, int act) {
  int lane = threadIdx.x, l15 = lane & 15, hi = lane >> 4;
  int m0 = blockIdx.x * 16, n0 = blockIdx.y * 16;
  const bf16* arow = A + (size_t)(m0 + l15) * K + (hi ? 16 : 0);
  const bf16* brow = Wt + (size_t)(n0 + l15) * K + (hi ? 16 : 0);
  v8f acc = {};
  for (int k0 = 0; k0 < K; k0 += 32) {
    v16bf a  = load_frag_A16(arow + k0);
    v16bf bb = load_frag_B16(brow + k0);
    acc = WMMA_BF16(a, bb, acc);
  }
  int n = n0 + l15;
  float bv = bias[n];
#pragma unroll
  for (int j = 0; j < 8; ++j) {
    int mm = m0 + j + hi * 8;
    float v = acc[j] + bv;
    v = (act == 0) ? fmaxf(v, 0.0f) : 1.0f / (1.0f + __expf(-v));
    if (outf)  outf[(size_t)mm * N + n]  = v;
    if (outbf) outbf[(size_t)mm * N + n] = (bf16)v;
  }
}

// ---------------------------------------------------------------------------
extern "C" void kernel_launch(void* const* d_in, const int* in_sizes, int n_in,
                              void* d_out, int out_size, void* d_ws, size_t ws_size,
                              hipStream_t stream) {
  const float* x   = (const float*)d_in[0];
  const int*   lab = (const int*)  d_in[1];
  const float* c1w = (const float*)d_in[2];
  const float* c1b = (const float*)d_in[3];
  const float* pw  = (const float*)d_in[4];
  const float* pb  = (const float*)d_in[5];
  const float* Wd  = (const float*)d_in[6];
  const float* w1  = (const float*)d_in[7];
  const float* b1  = (const float*)d_in[8];
  const float* w2  = (const float*)d_in[9];
  const float* b2  = (const float*)d_in[10];
  const float* w3  = (const float*)d_in[11];
  const float* b3  = (const float*)d_in[12];

  float* vout   = (float*)d_out;                 // [256,10,16]
  float* recon  = vout + 256 * 160;              // [256,784]
  float* onehot = vout + 256 * 160 + 256 * 784;  // [256,10]

  char* ws = (char*)d_ws;
  size_t off = 0;
  auto alloc = [&](size_t bytes) { char* p = ws + off; off += (bytes + 255) & ~(size_t)255; return p; };
  bf16*  w1b   = (bf16*) alloc((size_t)256 * 96 * 2);
  bf16*  A1    = (bf16*) alloc((size_t)102400 * 96 * 2);
  bf16*  h1    = (bf16*) alloc((size_t)256 * 400 * 256 * 2);
  bf16*  w2b   = (bf16*) alloc((size_t)256 * 20736 * 2);
  float* h2    = (float*)alloc((size_t)256 * 9216 * 4);
  float* caps  = (float*)alloc((size_t)256 * 9216 * 4);
  bf16*  recin = (bf16*) alloc((size_t)256 * 160 * 2);
  bf16*  wd1   = (bf16*) alloc((size_t)512 * 160 * 2);
  bf16*  wd2   = (bf16*) alloc((size_t)1024 * 512 * 2);
  bf16*  wd3   = (bf16*) alloc((size_t)784 * 1024 * 2);
  bf16*  a1    = (bf16*) alloc((size_t)256 * 512 * 2);
  bf16*  a2    = (bf16*) alloc((size_t)256 * 1024 * 2);

  // weight prep / precision conversion / conv1 im2col
  prep_w1<<<(256 * 96 + 255) / 256, 256, 0, stream>>>(c1w, w1b);
  prep_w2<<<(256 * 20736 + 255) / 256, 256, 0, stream>>>(pw, w2b);
  cvt_f32_bf16<<<(512 * 160 + 255) / 256, 256, 0, stream>>>(w1, wd1, 512 * 160);
  cvt_f32_bf16<<<(1024 * 512 + 255) / 256, 256, 0, stream>>>(w2, wd2, 1024 * 512);
  cvt_f32_bf16<<<(784 * 1024 + 255) / 256, 256, 0, stream>>>(w3, wd3, 784 * 1024);
  im2col1<<<(102400 * 96 + 255) / 256, 256, 0, stream>>>(x, A1);

  // conv stem + primary caps (WMMA bf16, f32 accumulate)
  conv1_wmma<<<dim3(102400 / 16, 256 / 16), 32, 0, stream>>>(A1, w1b, c1b, h1);
  pconv_wmma<<<dim3(9216 / 32, 256 / 32), 32, 0, stream>>>(h1, w2b, pb, h2);
  squash_caps<<<(256 * 1152 + 255) / 256, 256, 0, stream>>>(h2, caps);

  // dynamic routing: one block per (b, out-capsule); u tile in LDS
  size_t smem = (size_t)(1152 * U_STRIDE + 1152 + 1152 + 256 + 16 + 16) * 4;
  routing_kernel<<<2560, 256, smem, stream>>>(caps, Wd, vout);

  // masked reconstruction (decoder WMMA GEMMs)
  mask_onehot<<<(256 * 160 + 255) / 256, 256, 0, stream>>>(vout, lab, recin, onehot);
  gemm_dec<<<dim3(16, 512 / 16), 32, 0, stream>>>(recin, wd1, b1, nullptr, a1, 160, 512, 0);
  gemm_dec<<<dim3(16, 1024 / 16), 32, 0, stream>>>(a1, wd2, b2, nullptr, a2, 512, 1024, 0);
  gemm_dec<<<dim3(16, 784 / 16), 32, 0, stream>>>(a2, wd3, b3, recon, nullptr, 1024, 784, 1);
}
